// CPSFMemcellAutoencoder_44968307589598
// MI455X (gfx1250) — compile-verified, matrix operand order checked by reading, not run on hardware
//
#include <hip/hip_runtime.h>
#include <cstdint>

typedef __attribute__((ext_vector_type(16))) _Float16 v16h;
typedef __attribute__((ext_vector_type(8)))  _Float16 v8h;
typedef __attribute__((ext_vector_type(8)))  float    v8f;
typedef __attribute__((ext_vector_type(4)))  unsigned int u32x4;
typedef __attribute__((ext_vector_type(8)))  int i32x8;
typedef __attribute__((ext_vector_type(4)))  int i32x4;

#if defined(__gfx1250__) && __has_builtin(__builtin_amdgcn_tensor_load_to_lds) && \
    __has_builtin(__builtin_amdgcn_s_wait_tensorcnt)
#define ATHENA_HAS_TDM 1
#else
#define ATHENA_HAS_TDM 0
#endif

union AF { v16h v; v8h h[2]; };

// Per-lane k index for a 16-bit 16x32 A fragment (and, by A<->B symmetry, the
// k index of the B fragment column held by this lane). From ISA 7.12.2:
// VGPR0..3 hold K=(0..7)+8*lanehi, VGPR4..7 hold K=16+(0..7)+8*lanehi.
__device__ __forceinline__ int frag_k(int lane, int h) {
  return (h & 7) | ((h & 8) << 1) | ((lane & 16) >> 1);
}

__device__ __forceinline__ v8h h8_zero() {
  v8h r;
#pragma unroll
  for (int i = 0; i < 8; ++i) r[i] = (_Float16)0.f;
  return r;
}
__device__ __forceinline__ v8f f8_zero() {
  v8f r;
#pragma unroll
  for (int i = 0; i < 8; ++i) r[i] = 0.f;
  return r;
}
__device__ __forceinline__ float silu_f(float x) { return x / (1.f + __expf(-x)); }

__device__ __forceinline__ v8f wmma_f16(v16h a, v16h b, v8f c) {
  return __builtin_amdgcn_wmma_f32_16x16x32_f16(false, a, false, b, (short)0, c,
                                                false, false);
}

// ---------------------------------------------------------------------------
// Tensor Data Mover staging (1-D tile, 8-byte elements), with cooperative-copy
// fallback when the TDM builtin is unavailable.
// ---------------------------------------------------------------------------

__device__ __forceinline__ void tdm_load_1d(const void* gsrc, void* lds_dst,
                                            unsigned bytes) {
#if ATHENA_HAS_TDM
  unsigned long long ga = (unsigned long long)(uintptr_t)gsrc;
  // Flat LDS aperture: addr[31:0] is the LDS byte offset (ISA 10.2).
  unsigned lds = (unsigned)(uintptr_t)lds_dst;
  unsigned elems = bytes >> 3;  // 8-byte elements
  u32x4 g0;
  g0[0] = 1u;                                   // count=1, user descriptor
  g0[1] = lds;                                  // lds_addr[31:0]
  g0[2] = (unsigned)ga;                         // global_addr[31:0]
  g0[3] = (unsigned)((ga >> 32) & 0x1FFFFFFull) // global_addr[56:32]
          | (2u << 30);                         // type=2 ("image")
  i32x8 g1;
  g1[0] = (3 << 16);                            // wg_mask=0, data_size=8B
  g1[1] = (int)((elems & 0xFFFFu) << 16);       // tensor_dim0[15:0]
  g1[2] = (int)(((elems >> 16) & 0xFFFFu)       // tensor_dim0[31:16]
                | (1u << 16));                  // tensor_dim1 = 1
  g1[3] = (int)((elems & 0xFFFFu) << 16);       // tile_dim0 = elems
  g1[4] = 0;                                    // tile_dim1/2 unused
  g1[5] = (int)elems;                           // tensor_dim0_stride
  g1[6] = 0;
  g1[7] = 0;
  i32x4 gz;
  gz[0] = gz[1] = gz[2] = gz[3] = 0;
  i32x8 gz8;
#pragma unroll
  for (int i = 0; i < 8; ++i) gz8[i] = 0;
  __builtin_amdgcn_tensor_load_to_lds(g0, g1, gz, gz, gz8, 0);
#else
  (void)gsrc; (void)lds_dst; (void)bytes;
#endif
}

// Wave 0 issues the DMA (TDM path) or all threads copy (fallback).
__device__ __forceinline__ void stage_b(const _Float16* g, _Float16* s, int tid,
                                        int bytes) {
#if ATHENA_HAS_TDM
  if ((tid >> 5) == 0) tdm_load_1d(g, s, (unsigned)bytes);
#else
  const uint4* src = (const uint4*)g;
  uint4* dst = (uint4*)s;
  for (int i = tid; i < bytes / 16; i += 256) dst[i] = src[i];
#endif
}
__device__ __forceinline__ void stage_commit(int tid) {
#if ATHENA_HAS_TDM
  if ((tid >> 5) == 0) __builtin_amdgcn_s_wait_tensorcnt(0);
#endif
  __syncthreads();
}

// ---------------------------------------------------------------------------
// Pack kernels
// ---------------------------------------------------------------------------

// x [B,3,H,W] f32 NCHW -> xh [B,H,W,4] f16 (channel-padded NHWC)
__global__ void k_pack_x(const float* __restrict__ x, _Float16* __restrict__ xh,
                         int npix, int HW) {
  int gid = blockIdx.x * blockDim.x + threadIdx.x;
  if (gid >= npix) return;
  int b = gid / HW;
  int pix = gid - b * HW;
  const float* xb = x + (size_t)b * 3 * HW + pix;
  _Float16* o = xh + (size_t)gid * 4;
  o[0] = (_Float16)xb[0];
  o[1] = (_Float16)xb[HW];
  o[2] = (_Float16)xb[2 * HW];
  o[3] = (_Float16)0.f;
}

// Conv weights OIHW f32 -> WMMA B-fragment order f16.
// k = (dy*3+dx)*CIN + c ; layout [kb][nb][lane][16 halves]
__global__ void k_pack_convw(const float* __restrict__ wsrc,
                             _Float16* __restrict__ dst, int CIN, int COUT) {
  int K = CIN * 9;
  int KPAD = (K + 31) & ~31;
  int NB = COUT >> 4;
  int total = (KPAD >> 5) * NB * 512;
  int gid = blockIdx.x * blockDim.x + threadIdx.x;
  if (gid >= total) return;
  int h = gid & 15, lane = (gid >> 4) & 31, cell = gid >> 9;
  int nb = cell % NB, kb = cell / NB;
  int k = kb * 32 + frag_k(lane, h);
  int n = nb * 16 + (lane & 15);
  float v = 0.f;
  if (k < K) {
    int tap = k / CIN, c = k - tap * CIN;
    int dy = tap / 3, dx = tap - dy * 3;
    v = wsrc[((size_t)(n * CIN + c) * 3 + dy) * 3 + dx];
  }
  dst[gid] = (_Float16)v;
}

// Row-major [Krows][N] f32 -> B-fragment order f16 (cell_v)
__global__ void k_pack_mat(const float* __restrict__ src,
                           _Float16* __restrict__ dst, int Krows, int N) {
  int NB = N >> 4;
  int total = (((Krows + 31) >> 5)) * NB * 512;
  int gid = blockIdx.x * blockDim.x + threadIdx.x;
  if (gid >= total) return;
  int h = gid & 15, lane = (gid >> 4) & 31, cell = gid >> 9;
  int nb = cell % NB, kb = cell / NB;
  int k = kb * 32 + frag_k(lane, h);
  int n = nb * 16 + (lane & 15);
  float v = (k < Krows) ? src[(size_t)k * N + n] : 0.f;
  dst[gid] = (_Float16)v;
}

// Deconv weights [128][3][4][4] f32 -> [ky][kx][c][oc(pad4)] f16
__global__ void k_pack_dw(const float* __restrict__ dw,
                          _Float16* __restrict__ dst) {
  int gid = blockIdx.x * blockDim.x + threadIdx.x;  // 8192
  if (gid >= 8192) return;
  int oc = gid & 3, c = (gid >> 2) & 127, kx = (gid >> 9) & 3, ky = gid >> 11;
  float v = 0.f;
  if (oc < 3) v = dw[((size_t)(c * 3 + oc) * 4 + ky) * 4 + kx];
  dst[gid] = (_Float16)v;
}

// ---------------------------------------------------------------------------
// WMMA implicit-GEMM convolutions
// ---------------------------------------------------------------------------

// 3x3 stride-1 conv, CIN=3 (K=27 pad 32, one K-step) via LDS-staged im2col.
// in: [B,H,W,4] f16, out: [M=B*H*W, COUT] f16 NHWC with SiLU.
template <int COUT>
__global__ __launch_bounds__(256) void k_conv3(const _Float16* __restrict__ in,
                                               const _Float16* __restrict__ bp,
                                               const float* __restrict__ bias,
                                               _Float16* __restrict__ out,
                                               int H, int W) {
  constexpr int NB = COUT / 16;
  __shared__ _Float16 At[128][32];
  int tid = threadIdx.x, lane = tid & 31, wid = tid >> 5;
  int lanehi = lane >> 4, mlow = lane & 15;
  int tileBase = blockIdx.x * 128;

  for (int i = tid; i < 4096; i += 256) {
    int row = i >> 5, k = i & 31;
    _Float16 v = (_Float16)0.f;
    if (k < 27) {
      int tok = tileBase + row;
      int x = tok % W;
      int t2 = tok / W;
      int y = t2 % H;
      int b = t2 / H;
      int tap = k / 3, c = k - tap * 3;
      int dy = tap / 3, dx = tap - dy * 3;
      int iy = y + dy - 1, ix = x + dx - 1;
      if ((unsigned)iy < (unsigned)H && (unsigned)ix < (unsigned)W)
        v = in[((size_t)((b * H + iy) * W) + ix) * 4 + c];
    }
    At[row][k] = v;
  }
  __syncthreads();

  AF a;
  a.h[0] = *(const v8h*)&At[wid * 16 + mlow][lanehi * 8];
  a.h[1] = *(const v8h*)&At[wid * 16 + mlow][16 + lanehi * 8];

  int rowBase = tileBase + wid * 16;
#pragma unroll
  for (int nb = 0; nb < NB; ++nb) {
    AF bf;
    const v8h* pb = (const v8h*)(bp + (size_t)nb * 512 + lane * 16);
    bf.h[0] = pb[0];
    bf.h[1] = pb[1];
    v8f acc = wmma_f16(a.v, bf.v, f8_zero());
#pragma unroll
    for (int r = 0; r < 8; ++r) {
      int m = r + lanehi * 8;
      int n = nb * 16 + mlow;
      float v = acc[r] + bias[n];
      out[(size_t)(rowBase + m) * COUT + n] = (_Float16)silu_f(v);
    }
  }
}

// 3x3 stride-2 conv. A fragments load direct from NHWC f16 global (8-aligned
// k-runs stay inside one pixel's channel block -> contiguous 16B loads).
// For COUT=128 the packed B tile for each K-step is DMA'd into a
// double-buffered LDS tile by the Tensor Data Mover.
template <int CIN, int COUT>
__global__ __launch_bounds__(256) void k_conv_s2(
    const _Float16* __restrict__ in, const _Float16* __restrict__ bp,
    const float* __restrict__ bias, _Float16* __restrict__ out, int IH,
    int IW) {
  constexpr int NB = COUT / 16;
  constexpr int K = CIN * 9;
  constexpr int KPAD = (K + 31) & ~31;
  constexpr int KB = KPAD / 32;
  __shared__ _Float16 Bsh[2][NB * 512];
  int tid = threadIdx.x, lane = tid & 31, wid = tid >> 5;
  int lanehi = lane >> 4, mlow = lane & 15;
  int OH = IH >> 1, OW = IW >> 1;
  int rowBase = blockIdx.x * 128 + wid * 16;
  int tok = rowBase + mlow;
  int x = tok % OW;
  int t2 = tok / OW;
  int y = t2 % OH;
  int b = t2 / OH;
  int iy0 = 2 * y - 1, ix0 = 2 * x - 1;
  const _Float16* inb = in + (size_t)b * IH * IW * CIN;

  auto load_a = [&](int k0) -> AF {
    AF a;
    int ka = k0 + lanehi * 8;
#pragma unroll
    for (int half = 0; half < 2; ++half) {
      int k = ka + half * 16;
      v8h r = h8_zero();
      if (k < K) {
        int tap = k / CIN;
        int c = k - tap * CIN;
        int dy = tap / 3, dx = tap - dy * 3;
        int iy = iy0 + dy, ix = ix0 + dx;
        if ((unsigned)iy < (unsigned)IH && (unsigned)ix < (unsigned)IW)
          r = *(const v8h*)(inb + (size_t)(iy * IW + ix) * CIN + c);
      }
      a.h[half] = r;
    }
    return a;
  };

  v8f acc[NB];
#pragma unroll
  for (int nb = 0; nb < NB; ++nb) acc[nb] = f8_zero();

  if (NB == 8) {
    // --- TDM double-buffered B staging through LDS ---
    stage_b(bp, &Bsh[0][0], tid, NB * 1024);
    stage_commit(tid);
    for (int kb = 0; kb < KB; ++kb) {
      int cur = kb & 1;
      if (kb + 1 < KB)
        stage_b(bp + (size_t)(kb + 1) * NB * 512, &Bsh[cur ^ 1][0], tid,
                NB * 1024);
      AF a = load_a(kb * 32);
      const _Float16* bls = &Bsh[cur][0] + lane * 16;
#pragma unroll
      for (int nb = 0; nb < NB; ++nb) {
        AF bf;
        const v8h* pb = (const v8h*)(bls + nb * 512);
        bf.h[0] = pb[0];
        bf.h[1] = pb[1];
        acc[nb] = wmma_f16(a.v, bf.v, acc[nb]);
      }
      stage_commit(tid);
    }
  } else {
    // --- direct-from-global B fragments (tiny weights, L2 resident) ---
    const _Float16* bpl = bp + lane * 16;
    for (int k0 = 0; k0 < KPAD; k0 += 32) {
      AF a = load_a(k0);
      const _Float16* bpc = bpl + (size_t)(k0 >> 5) * NB * 512;
      if (k0 + 32 < KPAD) __builtin_prefetch(bpc + NB * 512, 0, 0);
#pragma unroll
      for (int nb = 0; nb < NB; ++nb) {
        AF bf;
        const v8h* pb = (const v8h*)(bpc + (size_t)nb * 512);
        bf.h[0] = pb[0];
        bf.h[1] = pb[1];
        acc[nb] = wmma_f16(a.v, bf.v, acc[nb]);
      }
    }
  }

#pragma unroll
  for (int nb = 0; nb < NB; ++nb)
#pragma unroll
    for (int r = 0; r < 8; ++r) {
      int m = r + lanehi * 8;
      int n = nb * 16 + mlow;
      float v = acc[nb][r] + bias[n];
      out[(size_t)(rowBase + m) * COUT + n] = (_Float16)silu_f(v);
    }
}

// ---------------------------------------------------------------------------
// Memcell
// ---------------------------------------------------------------------------

// w = softmax(z @ cell_k^T / 4): one thread per token (tiny GEMM, VALU).
__global__ __launch_bounds__(256) void k_w(const _Float16* __restrict__ z,
                                           const float* __restrict__ cell_k,
                                           _Float16* __restrict__ wout) {
  __shared__ float ck[512];
  int tid = threadIdx.x;
  for (int i = tid; i < 512; i += 256) ck[i] = cell_k[i];
  __syncthreads();
  int tok = blockIdx.x * 256 + tid;
  float zf[16];
  const _Float16* zp = z + (size_t)tok * 16;
#pragma unroll
  for (int j = 0; j < 16; ++j) zf[j] = (float)zp[j];
  float s[32];
  float mx = -1e30f;
#pragma unroll
  for (int m = 0; m < 32; ++m) {
    float d = 0.f;
#pragma unroll
    for (int j = 0; j < 16; ++j) d += zf[j] * ck[m * 16 + j];
    d *= 0.25f;  // 1/sqrt(16)
    s[m] = d;
    mx = fmaxf(mx, d);
  }
  float sum = 0.f;
#pragma unroll
  for (int m = 0; m < 32; ++m) {
    float e = __expf(s[m] - mx);
    s[m] = e;
    sum += e;
  }
  float inv = 1.f / sum;
  _Float16* wp = wout + (size_t)tok * 32;
#pragma unroll
  for (int m = 0; m < 32; ++m) wp[m] = (_Float16)(s[m] * inv);
}

// t_hat = w @ cell_v via WMMA; diff = t_star - t_hat staged in LDS; then
// per-block partial of w^T @ diff accumulated in registers.
__global__ __launch_bounds__(256) void k_memcell(
    const _Float16* __restrict__ wmat, const _Float16* __restrict__ cvp,
    const _Float16* __restrict__ tstar, float* __restrict__ partial,
    int tilesPerBlk) {
  __shared__ _Float16 diffS[128][128];
  __shared__ _Float16 wS[128][32];
  int tid = threadIdx.x, lane = tid & 31, wid = tid >> 5;
  int lanehi = lane >> 4, mlow = lane & 15;
  int rm = tid & 31;
  int rs0 = (tid >> 5) << 4;
  float racc[16];
#pragma unroll
  for (int j = 0; j < 16; ++j) racc[j] = 0.f;

  for (int it = 0; it < tilesPerBlk; ++it) {
    int tileBase = (blockIdx.x * tilesPerBlk + it) * 128;
    int rowA = tileBase + wid * 16 + mlow;
    AF a;
    a.h[0] = *(const v8h*)(wmat + (size_t)rowA * 32 + lanehi * 8);
    a.h[1] = *(const v8h*)(wmat + (size_t)rowA * 32 + 16 + lanehi * 8);
    {  // stage w tile for the reduction
      const uint4* src = (const uint4*)(wmat + (size_t)tileBase * 32);
      uint4* dst = (uint4*)&wS[0][0];
      for (int i = tid; i < 512; i += 256) dst[i] = src[i];
    }
#pragma unroll
    for (int nb = 0; nb < 8; ++nb) {
      AF bf;
      const v8h* pb = (const v8h*)(cvp + (size_t)nb * 512 + lane * 16);
      bf.h[0] = pb[0];
      bf.h[1] = pb[1];
      v8f th = wmma_f16(a.v, bf.v, f8_zero());
#pragma unroll
      for (int r = 0; r < 8; ++r) {
        int m = r + lanehi * 8;
        int n = nb * 16 + mlow;
        int trow = tileBase + wid * 16 + m;
        float d = (float)tstar[(size_t)trow * 128 + n] - th[r];
        diffS[wid * 16 + m][n] = (_Float16)d;
      }
    }
    __syncthreads();
    for (int t2 = 0; t2 < 128; ++t2) {
      float wv = (float)wS[t2][rm];
#pragma unroll
      for (int j = 0; j < 16; ++j)
        racc[j] += wv * (float)diffS[t2][rs0 + j];
    }
    __syncthreads();
  }
  float* po = partial + ((size_t)blockIdx.x * 32 + rm) * 128 + rs0;
#pragma unroll
  for (int j = 0; j < 16; ++j) po[j] = racc[j];
}

// cell_v_new = cell_v + ALPHA * sum(partials), emitted directly in B-fragment
// order (thread index == fragment half index: bijection onto 32x128).
__global__ void k_vreduce(const float* __restrict__ partial,
                          const float* __restrict__ cell_v,
                          _Float16* __restrict__ cvnp, int nblk) {
  int gid = blockIdx.x * blockDim.x + threadIdx.x;  // 4096
  int h = gid & 15, lane = (gid >> 4) & 31, nb = gid >> 9;
  int k = frag_k(lane, h);
  int n = nb * 16 + (lane & 15);
  float s = 0.f;
  for (int b2 = 0; b2 < nblk; ++b2)
    s += partial[((size_t)b2 * 32 + k) * 128 + n];
  cvnp[gid] = (_Float16)(cell_v[k * 128 + n] + 1e-6f * s);
}

// t_read = w @ cell_v_new (WMMA), token layout == NHWC d0_m.
__global__ __launch_bounds__(256) void k_tread(
    const _Float16* __restrict__ wmat, const _Float16* __restrict__ cvp,
    _Float16* __restrict__ tread) {
  int tid = threadIdx.x, lane = tid & 31, wid = tid >> 5;
  int lanehi = lane >> 4, mlow = lane & 15;
  int rowBase = blockIdx.x * 128 + wid * 16;
  int rowA = rowBase + mlow;
  AF a;
  a.h[0] = *(const v8h*)(wmat + (size_t)rowA * 32 + lanehi * 8);
  a.h[1] = *(const v8h*)(wmat + (size_t)rowA * 32 + 16 + lanehi * 8);
#pragma unroll
  for (int nb = 0; nb < 8; ++nb) {
    AF bf;
    const v8h* pb = (const v8h*)(cvp + (size_t)nb * 512 + lane * 16);
    bf.h[0] = pb[0];
    bf.h[1] = pb[1];
    v8f acc = wmma_f16(a.v, bf.v, f8_zero());
#pragma unroll
    for (int r = 0; r < 8; ++r) {
      int m = r + lanehi * 8;
      int n = nb * 16 + mlow;
      tread[(size_t)(rowBase + m) * 128 + n] = (_Float16)acc[r];
    }
  }
}

// ---------------------------------------------------------------------------
// Decoder
// ---------------------------------------------------------------------------

// ConvTranspose2d(128->3, k=4, s=2, p=1) + bias + SiLU. One thread per output
// pixel; exactly 2x2 taps contribute per output parity.
__global__ __launch_bounds__(256) void k_deconv(
    const _Float16* __restrict__ dm, const _Float16* __restrict__ dwp,
    const float* __restrict__ db, _Float16* __restrict__ d1) {
  int gid = blockIdx.x * 256 + threadIdx.x;  // B*256*256
  int x = gid & 255, y = (gid >> 8) & 255, b = gid >> 16;
  float a0 = db[0], a1 = db[1], a2 = db[2];
  int py = (y + 1) & 1, px = (x + 1) & 1;
#pragma unroll
  for (int ii = 0; ii < 2; ++ii) {
    int ky = py + 2 * ii;
    int iy = (y + 1 - ky) >> 1;
    if ((unsigned)iy >= 128u) continue;
#pragma unroll
    for (int jj = 0; jj < 2; ++jj) {
      int kx = px + 2 * jj;
      int ix = (x + 1 - kx) >> 1;
      if ((unsigned)ix >= 128u) continue;
      const _Float16* ip = dm + ((size_t)((b * 128 + iy) * 128) + ix) * 128;
      const _Float16* wp = dwp + (size_t)(ky * 4 + kx) * 512;
      for (int c0 = 0; c0 < 128; c0 += 8) {
        v8h iv = *(const v8h*)(ip + c0);
#pragma unroll
        for (int j = 0; j < 8; ++j) {
          float v = (float)iv[j];
          const _Float16* w4 = wp + (c0 + j) * 4;
          a0 += v * (float)w4[0];
          a1 += v * (float)w4[1];
          a2 += v * (float)w4[2];
        }
      }
    }
  }
  _Float16* o = d1 + (size_t)gid * 4;
  o[0] = (_Float16)silu_f(a0);
  o[1] = (_Float16)silu_f(a1);
  o[2] = (_Float16)silu_f(a2);
  o[3] = (_Float16)0.f;
}

// Final 3x3 conv 3->3 + bias + SiLU, emit f32 NCHW to d_out.
__global__ __launch_bounds__(256) void k_final(const _Float16* __restrict__ d1,
                                               const float* __restrict__ cw,
                                               const float* __restrict__ cb,
                                               float* __restrict__ outp) {
  __shared__ float cwS[81];
  __shared__ float cbS[3];
  int tid = threadIdx.x;
  if (tid < 81) cwS[tid] = cw[tid];
  if (tid < 3) cbS[tid] = cb[tid];
  __syncthreads();
  int gid = blockIdx.x * 256 + tid;
  int x = gid & 255, y = (gid >> 8) & 255, b = gid >> 16;
  float acc0 = cbS[0], acc1 = cbS[1], acc2 = cbS[2];
#pragma unroll
  for (int dy = 0; dy < 3; ++dy) {
    int iy = y + dy - 1;
    if ((unsigned)iy >= 256u) continue;
#pragma unroll
    for (int dx = 0; dx < 3; ++dx) {
      int ix = x + dx - 1;
      if ((unsigned)ix >= 256u) continue;
      const _Float16* p = d1 + ((size_t)((b * 256 + iy) * 256) + ix) * 4;
#pragma unroll
      for (int ic = 0; ic < 3; ++ic) {
        float v = (float)p[ic];
        int t = ic * 9 + dy * 3 + dx;
        acc0 += v * cwS[t];
        acc1 += v * cwS[27 + t];
        acc2 += v * cwS[54 + t];
      }
    }
  }
  size_t plane = 65536;
  size_t ob = (size_t)b * 3 * plane + (size_t)y * 256 + x;
  outp[ob] = silu_f(acc0);
  outp[ob + plane] = silu_f(acc1);
  outp[ob + 2 * plane] = silu_f(acc2);
}

// ---------------------------------------------------------------------------

extern "C" void kernel_launch(void* const* d_in, const int* in_sizes, int n_in,
                              void* d_out, int out_size, void* d_ws,
                              size_t ws_size, hipStream_t stream) {
  (void)in_sizes; (void)n_in; (void)out_size; (void)ws_size;
  const float* x      = (const float*)d_in[0];
  const float* e0n_w1 = (const float*)d_in[1];
  const float* e0n_b1 = (const float*)d_in[2];
  const float* e0n_w2 = (const float*)d_in[3];
  const float* e0n_b2 = (const float*)d_in[4];
  const float* e0s_w1 = (const float*)d_in[5];
  const float* e0s_b1 = (const float*)d_in[6];
  const float* e0s_w2 = (const float*)d_in[7];
  const float* e0s_b2 = (const float*)d_in[8];
  const float* d0_dw  = (const float*)d_in[9];
  const float* d0_db  = (const float*)d_in[10];
  const float* d0_cw  = (const float*)d_in[11];
  const float* d0_cb  = (const float*)d_in[12];
  const float* cell_k = (const float*)d_in[13];
  const float* cell_v = (const float*)d_in[14];

  const int B = 8, H = 256, W = 256, HW = H * W;
  const int T = B * (H / 2) * (W / 2);  // 131072 tokens

  size_t off = 0;
  auto carve = [&](size_t bytes) -> char* {
    char* p = (char*)d_ws + off;
    off += (bytes + 255) & ~(size_t)255;
    return p;
  };
  _Float16* xh    = (_Float16*)carve((size_t)B * HW * 4 * 2);
  _Float16* actS  = (_Float16*)carve((size_t)B * HW * 128 * 2);
  _Float16* actN  = (_Float16*)carve((size_t)B * HW * 16 * 2);
  _Float16* tstar = (_Float16*)carve((size_t)T * 128 * 2);
  _Float16* zbuf  = (_Float16*)carve((size_t)T * 16 * 2);
  _Float16* wbuf  = (_Float16*)carve((size_t)T * 32 * 2);
  _Float16* tread = (_Float16*)carve((size_t)T * 128 * 2);
  _Float16* d1    = (_Float16*)carve((size_t)B * HW * 4 * 2);
  _Float16* bpS1  = (_Float16*)carve((size_t)1 * 8 * 512 * 2);
  _Float16* bpS2  = (_Float16*)carve((size_t)36 * 8 * 512 * 2);
  _Float16* bpN1  = (_Float16*)carve((size_t)1 * 1 * 512 * 2);
  _Float16* bpN2  = (_Float16*)carve((size_t)5 * 1 * 512 * 2);
  _Float16* cvp   = (_Float16*)carve((size_t)8 * 512 * 2);
  _Float16* cvnp  = (_Float16*)carve((size_t)8 * 512 * 2);
  _Float16* dwp   = (_Float16*)carve((size_t)8192 * 2);
  float*    part  = (float*)carve((size_t)256 * 32 * 128 * 4);

  // --- pack inputs/weights ---
  k_pack_x<<<(B * HW + 255) / 256, 256, 0, stream>>>(x, xh, B * HW, HW);
  k_pack_convw<<<16, 256, 0, stream>>>(e0s_w1, bpS1, 3, 128);    // 4096
  k_pack_convw<<<576, 256, 0, stream>>>(e0s_w2, bpS2, 128, 128); // 147456
  k_pack_convw<<<2, 256, 0, stream>>>(e0n_w1, bpN1, 3, 16);      // 512
  k_pack_convw<<<10, 256, 0, stream>>>(e0n_w2, bpN2, 16, 16);    // 2560
  k_pack_mat<<<16, 256, 0, stream>>>(cell_v, cvp, 32, 128);
  k_pack_dw<<<32, 256, 0, stream>>>(d0_dw, dwp);

  // --- encoders (WMMA implicit GEMM) ---
  k_conv3<128><<<(B * HW) / 128, 256, 0, stream>>>(xh, bpS1, e0s_b1, actS, H, W);
  k_conv_s2<128, 128><<<T / 128, 256, 0, stream>>>(actS, bpS2, e0s_b2, tstar, H, W);
  k_conv3<16><<<(B * HW) / 128, 256, 0, stream>>>(xh, bpN1, e0n_b1, actN, H, W);
  k_conv_s2<16, 16><<<T / 128, 256, 0, stream>>>(actN, bpN2, e0n_b2, zbuf, H, W);

  // --- memcell ---
  k_w<<<T / 256, 256, 0, stream>>>(zbuf, cell_k, wbuf);
  k_memcell<<<256, 256, 0, stream>>>(wbuf, cvp, tstar, part, (T / 128) / 256);
  k_vreduce<<<16, 256, 0, stream>>>(part, cell_v, cvnp, 256);
  k_tread<<<T / 128, 256, 0, stream>>>(wbuf, cvnp, tread);

  // --- decoder ---
  k_deconv<<<(B * HW) / 256, 256, 0, stream>>>(tread, dwp, d0_db, d1);
  k_final<<<(B * HW) / 256, 256, 0, stream>>>(d1, d0_cw, d0_cb, (float*)d_out);
}
